// JetEdgeClassifier_72516227826286
// MI455X (gfx1250) — matrix operand order repaired and checked by reading, not compiled, not run on hardware
//
#include <hip/hip_runtime.h>

typedef unsigned short u16;
typedef unsigned int   u32;
typedef __attribute__((ext_vector_type(16))) __bf16 v16bf;
typedef __attribute__((ext_vector_type(8)))  float  v8f;
typedef __attribute__((ext_vector_type(4)))  u32    u32x4;

#define ASTRIDE 320   // activation row stride (bf16 elements), >= max padded width (320)

__device__ __forceinline__ u16 f2bf(float f) {
    u32 u = __float_as_uint(f);
    u += 0x7FFFu + ((u >> 16) & 1u);   // round-to-nearest-even
    return (u16)(u >> 16);
}

union ABu { v16bf v; u32x4 q[2]; };

// One padded MLP layer for TWO 16-row tiles held in LDS (M=32 per wave).
// B fragments are loaded once per k-tile and reused by two independent WMMAs.
// If gOut != nullptr this is a final layer: write acc+bias (f32) to global.
// act: 0 = none, 1 = relu, 2 = tanh.
__device__ __forceinline__ void mlp_layer2(
    const u16* __restrict__ Wp, const float* __restrict__ bias,
    const u16* in0, const u16* in1, u16* out0, u16* out1,
    int Kp, int Np, int act,
    float* gOut, int gStride, int gCols, int gRowBase)
{
    const int lane = threadIdx.x & 31;
    const int half = lane >> 4;
    const int n0   = lane & 15;
    const int ktiles = Kp >> 5;
    const int ntiles = Np >> 4;
    for (int nt = 0; nt < ntiles; ++nt) {
        v8f acc0 = {}, acc1 = {};
        const u16* wt  = Wp + (size_t)(nt * ktiles) * 512 + lane * 16;
        const u16* ar0 = in0 + n0 * ASTRIDE + half * 16;  // A row m = lane&15
        const u16* ar1 = in1 + n0 * ASTRIDE + half * 16;
        for (int kt = 0; kt < ktiles; ++kt) {
            ABu a0, a1, b;
            b.q[0]  = *(const u32x4*)(wt);         // packed: lane-contiguous 16 bf16
            b.q[1]  = *(const u32x4*)(wt + 8);
            a0.q[0] = *(const u32x4*)(ar0);        // K = kt*32 + half*16 + 0..7
            a0.q[1] = *(const u32x4*)(ar0 + 8);    // K = kt*32 + half*16 + 8..15
            a1.q[0] = *(const u32x4*)(ar1);
            a1.q[1] = *(const u32x4*)(ar1 + 8);
            acc0 = __builtin_amdgcn_wmma_f32_16x16x32_bf16(
                       false, a0.v, false, b.v, (short)0, acc0, false, false);
            acc1 = __builtin_amdgcn_wmma_f32_16x16x32_bf16(
                       false, a1.v, false, b.v, (short)0, acc1, false, false);
            ar0 += 32; ar1 += 32; wt += 512;
        }
        float bn = bias[nt * 16 + n0];
        if (gOut) {
            if (n0 < gCols) {
#pragma unroll
                for (int r = 0; r < 8; ++r) {
                    int m = r + half * 8;
                    gOut[(size_t)(gRowBase + m) * gStride + n0]      = acc0[r] + bn;
                    gOut[(size_t)(gRowBase + 16 + m) * gStride + n0] = acc1[r] + bn;
                }
            }
        } else {
#pragma unroll
            for (int r = 0; r < 8; ++r) {
                int m = r + half * 8;
                float v0 = acc0[r] + bn;
                float v1 = acc1[r] + bn;
                if (act == 1)      { v0 = fmaxf(v0, 0.f); v1 = fmaxf(v1, 0.f); }
                else if (act == 2) { v0 = tanhf(v0);      v1 = tanhf(v1); }
                out0[m * ASTRIDE + nt * 16 + n0] = f2bf(v0);
                out1[m * ASTRIDE + nt * 16 + n0] = f2bf(v1);
            }
        }
    }
    __syncthreads();
}

// ---- weight/bias repack to padded bf16 WMMA tile order ----
__global__ void pack_w(const float* __restrict__ W, u16* __restrict__ dst,
                       int fi, int fo, int Kp, int Np) {
    int idx = blockIdx.x * 256 + threadIdx.x;
    if (idx >= Kp * Np) return;
    int t = idx >> 9, e = idx & 511;
    int lane = e >> 4, i = e & 15;
    int ktiles = Kp >> 5;
    int kt = t % ktiles, nt = t / ktiles;
    int n = nt * 16 + (lane & 15);
    int k = kt * 32 + ((lane >> 4) << 4) + i;   // B layout: lanes 0-15 K=0..15, 16-31 K=16..31
    float v = (k < fi && n < fo) ? W[(size_t)k * fo + n] : 0.f;
    dst[idx] = f2bf(v);
}

__global__ void pack_b(const float* __restrict__ b, float* __restrict__ dst,
                       int fo, int Np) {
    int i = blockIdx.x * 64 + threadIdx.x;
    if (i < Np) dst[i] = (i < fo) ? b[i] : 0.f;
}

// ---- fused node_init + node_class, 32 nodes (two 16-row tiles) per wave ----
__global__ __launch_bounds__(32) void node_kernel(
    const float* __restrict__ jet, const float* __restrict__ nodef,
    const u16* __restrict__ Wp, const float* __restrict__ Bp,
    u16* __restrict__ hbf, float* __restrict__ outNC)
{
    __shared__ __align__(16) u16 A0[16 * ASTRIDE];
    __shared__ __align__(16) u16 A1[16 * ASTRIDE];
    __shared__ __align__(16) u16 B0[16 * ASTRIDE];
    __shared__ __align__(16) u16 B1[16 * ASTRIDE];
    const int lane = threadIdx.x;
    const int tile = blockIdx.x;          // 32 nodes per block
    const int t = lane >> 4, r = lane & 15;

    {   // input: [node(6), jet(2)] padded to 32; one row per lane
        int node = tile * 32 + lane;
        u16* row = (t ? A1 : A0) + r * ASTRIDE;
        for (int c = 0; c < 6; ++c) row[c] = f2bf(nodef[(size_t)node * 6 + c]);
        row[6] = f2bf(jet[(size_t)node * 2 + 0]);
        row[7] = f2bf(jet[(size_t)node * 2 + 1]);
        for (int c = 8; c < 32; ++c) row[c] = 0;
    }
    __syncthreads();

    // node_init: (8,100)(100,100)(100,50)(50,10)(10,128), tanh -> h in B0/B1
    mlp_layer2(Wp + 0,     Bp + 0,   A0, A1, B0, B1, 32, 128, 1, nullptr, 0, 0, 0);
    mlp_layer2(Wp + 4096,  Bp + 128, B0, B1, A0, A1, 128,128, 1, nullptr, 0, 0, 0);
    mlp_layer2(Wp + 20480, Bp + 256, A0, A1, B0, B1, 128, 64, 1, nullptr, 0, 0, 0);
    mlp_layer2(Wp + 28672, Bp + 320, B0, B1, A0, A1, 64,  32, 1, nullptr, 0, 0, 0);
    mlp_layer2(Wp + 30720, Bp + 352, A0, A1, B0, B1, 32, 128, 2, nullptr, 0, 0, 0);

    {   // store h (bf16) for edge kernel (L2-resident 25.6 MB) and
        // build node_class input [node(6), h(128), jet(2)] pad 160 into A
        int node = tile * 32 + lane;
        const u16* hrow = (t ? B1 : B0) + r * ASTRIDE;
        u32* d = (u32*)(hbf + (size_t)node * 128);
        const u32* s = (const u32*)hrow;
#pragma unroll
        for (int i = 0; i < 64; ++i) d[i] = s[i];

        u16* row = (t ? A1 : A0) + r * ASTRIDE;
        for (int c = 0; c < 6; ++c) row[c] = f2bf(nodef[(size_t)node * 6 + c]);
        u32* dl = (u32*)(row + 6);
        for (int i = 0; i < 64; ++i) dl[i] = s[i];
        row[134] = f2bf(jet[(size_t)node * 2 + 0]);
        row[135] = f2bf(jet[(size_t)node * 2 + 1]);
        for (int c = 136; c < 160; ++c) row[c] = 0;
    }
    __syncthreads();

    // node_class: (136,300)(300,200)(200,100)(100,2), final -> global f32
    mlp_layer2(Wp + 34816,  Bp + 480,  A0, A1, B0, B1, 160, 320, 1, nullptr, 0, 0, 0);
    mlp_layer2(Wp + 86016,  Bp + 800,  B0, B1, A0, A1, 320, 224, 1, nullptr, 0, 0, 0);
    mlp_layer2(Wp + 157696, Bp + 1024, A0, A1, B0, B1, 224, 128, 1, nullptr, 0, 0, 0);
    mlp_layer2(Wp + 186368, Bp + 1152, B0, B1, nullptr, nullptr, 128, 16, 0,
               outNC, 2, 2, tile * 32);
}

// ---- edge_class, 32 edges (two 16-row tiles) per wave ----
__global__ __launch_bounds__(32) void edge_kernel(
    const float* __restrict__ jet, const float* __restrict__ nodef,
    const int* __restrict__ srcI, const int* __restrict__ dstI,
    const u16* __restrict__ Wp, const float* __restrict__ Bp,
    const u16* __restrict__ hbf, float* __restrict__ outE)
{
    __shared__ __align__(16) u16 A0[16 * ASTRIDE];
    __shared__ __align__(16) u16 A1[16 * ASTRIDE];
    __shared__ __align__(16) u16 B0[16 * ASTRIDE];
    __shared__ __align__(16) u16 B1[16 * ASTRIDE];
    const int lane = threadIdx.x;
    const int tile = blockIdx.x;          // 32 edges per block
    const int t = lane >> 4, r = lane & 15;

    {   // z = [jet_s(2), node_s(6), h_s(128), node_d(6), h_d(128)] = 270, pad 288
        int e  = tile * 32 + lane;
        int sn = srcI[e];
        int dn = dstI[e];
        u16* row = (t ? A1 : A0) + r * ASTRIDE;
        row[0] = f2bf(jet[(size_t)sn * 2 + 0]);
        row[1] = f2bf(jet[(size_t)sn * 2 + 1]);
        for (int c = 0; c < 6; ++c) row[2 + c] = f2bf(nodef[(size_t)sn * 6 + c]);
        u32* ds = (u32*)(row + 8);
        const u32* ss = (const u32*)(hbf + (size_t)sn * 128);
#pragma unroll
        for (int i = 0; i < 64; ++i) ds[i] = ss[i];
        for (int c = 0; c < 6; ++c) row[136 + c] = f2bf(nodef[(size_t)dn * 6 + c]);
        u32* dd = (u32*)(row + 142);
        const u32* sd = (const u32*)(hbf + (size_t)dn * 128);
#pragma unroll
        for (int i = 0; i < 64; ++i) dd[i] = sd[i];
        for (int c = 270; c < 288; ++c) row[c] = 0;
    }
    __syncthreads();

    // edge_class: (270,100)(100,100)(100,50)(50,20)(20,10)(10,1), final -> global f32
    mlp_layer2(Wp + 188416, Bp + 1168, A0, A1, B0, B1, 288, 128, 1, nullptr, 0, 0, 0);
    mlp_layer2(Wp + 225280, Bp + 1296, B0, B1, A0, A1, 128, 128, 1, nullptr, 0, 0, 0);
    mlp_layer2(Wp + 241664, Bp + 1424, A0, A1, B0, B1, 128,  64, 1, nullptr, 0, 0, 0);
    mlp_layer2(Wp + 249856, Bp + 1488, B0, B1, A0, A1,  64,  32, 1, nullptr, 0, 0, 0);
    mlp_layer2(Wp + 251904, Bp + 1520, A0, A1, B0, B1,  32,  32, 1, nullptr, 0, 0, 0);
    mlp_layer2(Wp + 252928, Bp + 1552, B0, B1, nullptr, nullptr, 32, 16, 0,
               outE, 1, 1, tile * 32);
}

extern "C" void kernel_launch(void* const* d_in, const int* in_sizes, int n_in,
                              void* d_out, int out_size, void* d_ws, size_t ws_size,
                              hipStream_t stream) {
    (void)in_sizes; (void)n_in; (void)out_size; (void)ws_size;
    const float* jet   = (const float*)d_in[0];
    const float* nodef = (const float*)d_in[1];
    const int*   srcI  = (const int*)d_in[2];
    const int*   dstI  = (const int*)d_in[3];

    // 15 layers: node_init(5), node_class(4), edge_class(6); padded K mult-32, N mult-16
    static const int Lfi[15] = {8,100,100,50,10, 136,300,200,100, 270,100,100,50,20,10};
    static const int Lfo[15] = {100,100,50,10,128, 300,200,100,2, 100,100,50,20,10,1};
    static const int LKp[15] = {32,128,128,64,32, 160,320,224,128, 288,128,128,64,32,32};
    static const int LNp[15] = {128,128,64,32,128, 320,224,128,16, 128,128,64,32,32,16};

    // d_ws layout: [0,506880) packed bf16 W; [506880,513152) padded f32 bias;
    // [513152, +25.6MB) bf16 h[100000][128]
    u16*   Wp  = (u16*)d_ws;
    float* Bp  = (float*)((char*)d_ws + 506880);
    u16*   hbf = (u16*)((char*)d_ws + 513152);

    size_t woff = 0, boff = 0;
    for (int l = 0; l < 15; ++l) {
        const float* W = (const float*)d_in[4 + 2 * l];
        const float* b = (const float*)d_in[5 + 2 * l];
        int total = LKp[l] * LNp[l];
        pack_w<<<(total + 255) / 256, 256, 0, stream>>>(W, Wp + woff,
                                                        Lfi[l], Lfo[l], LKp[l], LNp[l]);
        pack_b<<<(LNp[l] + 63) / 64, 64, 0, stream>>>(b, Bp + boff, Lfo[l], LNp[l]);
        woff += total; boff += LNp[l];
    }

    float* out = (float*)d_out;
    node_kernel<<<100000 / 32, 32, 0, stream>>>(jet, nodef, Wp, Bp, hbf, out);
    edge_kernel<<<1000000 / 32, 32, 0, stream>>>(jet, nodef, srcI, dstI, Wp, Bp, hbf,
                                                 out + 200000);
}
